// ATTENTION_13675175870765
// MI455X (gfx1250) — matrix operand order repaired and checked by reading, not compiled, not run on hardware
//
#include <hip/hip_runtime.h>

typedef __attribute__((ext_vector_type(16))) _Float16 v16h;
typedef __attribute__((ext_vector_type(8)))  _Float16 v8h;
typedef __attribute__((ext_vector_type(8)))  float    v8f;

#define ITEMNUM 50000
#define NEGINF  -3.0e38f

// ---------------------------------------------------------------------------
// prep: pe gather, seqs/time update, keep mask, gates = sigmoid(time . gate_W)
// one block (256 threads) per (b,l) row
// ---------------------------------------------------------------------------
__global__ __launch_bounds__(256) void prep_kernel(
    const int* __restrict__ seqs_data, const float* __restrict__ seqs,
    const int* __restrict__ position, const float* __restrict__ time_emb,
    const float* __restrict__ pos_table,
    const float* __restrict__ gate_W, const float* __restrict__ gate_b,
    float* __restrict__ s32, float* __restrict__ t32, _Float16* __restrict__ t16,
    float* __restrict__ gates, float* __restrict__ keep, int H, int NB)
{
    const int row = blockIdx.x, tid = threadIdx.x;
    __shared__ float red[256];
    const int pos = position[row];
    const float pe = pos_table[(size_t)pos * H + tid];
    float s = seqs[(size_t)row * H + tid] + pe;
    const float t = time_emb[(size_t)row * H + tid] + pe;
    const float kp = (seqs_data[row] != (ITEMNUM - 1)) ? 1.0f : 0.0f;
    s *= kp;
    s32[(size_t)row * H + tid] = s;
    t32[(size_t)row * H + tid] = t;
    t16[(size_t)row * H + tid] = (_Float16)t;
    if (tid == 0) keep[row] = kp;
    for (int n = 0; n < NB; ++n) {
        red[tid] = t * gate_W[(size_t)n * H + tid];
        __syncthreads();
        for (int st = 128; st > 0; st >>= 1) {
            if (tid < st) red[tid] += red[tid + st];
            __syncthreads();
        }
        if (tid == 0)
            gates[(size_t)row * NB + n] = 1.0f / (1.0f + __expf(-(red[0] + gate_b[n])));
        __syncthreads();
    }
}

// ---------------------------------------------------------------------------
// LayerNorm over H=256 (blockDim must equal H); optional residual add first
// ---------------------------------------------------------------------------
__global__ __launch_bounds__(256) void ln_kernel(
    const float* __restrict__ x, const float* __restrict__ res,
    const float* __restrict__ g, const float* __restrict__ bta,
    float* __restrict__ outF, _Float16* __restrict__ outH, int H)
{
    const int row = blockIdx.x, tid = threadIdx.x;
    __shared__ float red[256];
    float v = x[(size_t)row * H + tid];
    if (res) v += res[(size_t)row * H + tid];
    red[tid] = v;
    __syncthreads();
    for (int st = 128; st > 0; st >>= 1) {
        if (tid < st) red[tid] += red[tid + st];
        __syncthreads();
    }
    const float mean = red[0] / (float)H;
    __syncthreads();
    const float d = v - mean;
    red[tid] = d * d;
    __syncthreads();
    for (int st = 128; st > 0; st >>= 1) {
        if (tid < st) red[tid] += red[tid + st];
        __syncthreads();
    }
    const float var = red[0] / (float)H;
    __syncthreads();
    const float o = d * rsqrtf(var + 1e-8f) * g[tid] + bta[tid];
    if (outF) outF[(size_t)row * H + tid] = o;
    if (outH) outH[(size_t)row * H + tid] = (_Float16)o;
}

// ---------------------------------------------------------------------------
// elementwise helpers
// ---------------------------------------------------------------------------
__global__ void cvt_f32_f16(const float* __restrict__ a, _Float16* __restrict__ o, size_t n)
{
    size_t i = (size_t)blockIdx.x * blockDim.x + threadIdx.x;
    if (i < n) o[i] = (_Float16)a[i];
}

__global__ void add_f32_to_f16(const float* __restrict__ a, const float* __restrict__ b,
                               _Float16* __restrict__ o, size_t n)
{
    size_t i = (size_t)blockIdx.x * blockDim.x + threadIdx.x;
    if (i < n) o[i] = (_Float16)(a[i] + b[i]);
}

// v (B,L,H) -> vT (B,NH,HD,L)
__global__ void transpose_v_kernel(const _Float16* __restrict__ v, _Float16* __restrict__ vT,
                                   int B, int L, int NH, int HD)
{
    size_t i = (size_t)blockIdx.x * blockDim.x + threadIdx.x;
    size_t total = (size_t)B * L * NH * HD;
    if (i >= total) return;
    int l = (int)(i % L);
    size_t t = i / L;
    int d = (int)(t % HD); t /= HD;
    int h = (int)(t % NH); t /= NH;
    int b = (int)t;
    vT[i] = v[((size_t)(b * L + l)) * (NH * HD) + h * HD + d];
}

// ---------------------------------------------------------------------------
// GEMM: out[M,N] = A[M,K](f16) @ W[N,K]^T(f16) + bias  (+residual)(relu)(*keep)
// wave32: each wave owns a 16(M) x 64(N) tile, 4 waves/block -> 16 x 256 tile.
// The shared 16xK A tile is staged into LDS once per block with the CDNA5
// async global->LDS path (ASYNCcnt), so the VMEM pipe only streams W.
// ---------------------------------------------------------------------------
__global__ __launch_bounds__(128) void gemm16_kernel(
    const _Float16* __restrict__ A, const _Float16* __restrict__ W,
    const float* __restrict__ bias, const float* __restrict__ residual,
    const float* __restrict__ keep,
    float* __restrict__ outF, _Float16* __restrict__ outH,
    int M, int K, int N, int relu)
{
    const int lane = threadIdx.x;
    const int hi = lane >> 4, lo = lane & 15;
    const int m0 = blockIdx.x * 16;
    const int n0 = threadIdx.y * 64;

    __shared__ __align__(16) _Float16 Ash[16 * 256];   // K <= 256

    // --- cooperative async copy of the contiguous 16xK A tile into LDS ------
    {
        const int tid = threadIdx.y * 32 + threadIdx.x;          // 0..127
        const char* gbase = (const char*)(A + (size_t)m0 * K);   // 32*K bytes, contiguous
        const unsigned ldsbase = (unsigned)(size_t)(void*)Ash;   // addr[31:0] = LDS offset
        const int nchunks = 2 * K;                               // 16-byte chunks
        for (int c = tid; c < nchunks; c += 128) {
            const unsigned boff = (unsigned)c * 16u;
            asm volatile("global_load_async_to_lds_b128 %0, %1, off"
                         :: "v"(ldsbase + boff), "v"(gbase + boff)
                         : "memory");
        }
        asm volatile("s_wait_asynccnt 0x0" ::: "memory");        // wave-local completion
        __syncthreads();                                         // cross-wave visibility
    }

    v8f acc[4] = {};
    const _Float16* arow = &Ash[lo * K];

    for (int k0 = 0; k0 < K; k0 += 32) {
        // A tile 16x32 f16 from LDS, contiguous k order (not the ISA table order)
        // (lanes<16: K0-15, lanes>=16: K16-31, contiguous to match B)
        v8h alo = *(const v8h*)(arow + k0 + hi * 16);
        v8h ahi = *(const v8h*)(arow + k0 + hi * 16 + 8);
        v16h a;
        #pragma unroll
        for (int i = 0; i < 8; ++i) { a[i] = alo[i]; a[8 + i] = ahi[i]; }
        #pragma unroll
        for (int j = 0; j < 4; ++j) {
            // B tile 32x16: lane holds column n = n0+j*16+lo, halves K contiguous
            const _Float16* wp = W + (size_t)(n0 + j * 16 + lo) * K + k0 + hi * 16;
            v16h bb = *(const v16h*)wp;
            acc[j] = __builtin_amdgcn_wmma_f32_16x16x32_f16(
                false, a, false, bb, (short)0, acc[j], false, false);
        }
    }

    #pragma unroll
    for (int j = 0; j < 4; ++j) {
        const int col = n0 + j * 16 + lo;
        const float bv = bias ? bias[col] : 0.0f;
        #pragma unroll
        for (int r = 0; r < 8; ++r) {
            const int row = m0 + r + hi * 8;
            float v = acc[j][r] + bv;
            if (residual) v += residual[(size_t)row * N + col];
            if (relu)     v = fmaxf(v, 0.0f);
            if (keep)     v *= keep[row];
            if (outF) outF[(size_t)row * N + col] = v;
            if (outH) outH[(size_t)row * N + col] = (_Float16)v;
        }
    }
}

// ---------------------------------------------------------------------------
// Flash attention (causal, gated): one wave per (b, head, 16-query tile)
// S = (Q K^T) * scale * g_q * g_k  with causal mask, online softmax,
// P repacked C-layout -> A-layout via LDS, O += P V via WMMA.
// ---------------------------------------------------------------------------
__global__ __launch_bounds__(32) void attn_kernel(
    const _Float16* __restrict__ q, const _Float16* __restrict__ k,
    const _Float16* __restrict__ vT, const float* __restrict__ gates,
    float* __restrict__ out, int B, int L, int H, int NH, int HD, int NB,
    int blk, float scale)
{
    const int lane = threadIdx.x;
    const int hi = lane >> 4, lo = lane & 15;
    const int qt = blockIdx.x, h = blockIdx.y, b = blockIdx.z;
    const int q0 = qt * 16;
    __shared__ __align__(64) _Float16 Pst[16 * 32];

    // Q as two 16x32 A-tiles over hd=64
    const _Float16* qrow = q + ((size_t)(b * L + q0 + lo)) * H + h * HD;
    v16h aq[2];
    #pragma unroll
    for (int t = 0; t < 2; ++t) {
        v8h al = *(const v8h*)(qrow + t * 32 + hi * 16);
        v8h ah = *(const v8h*)(qrow + t * 32 + hi * 16 + 8);
        #pragma unroll
        for (int i = 0; i < 8; ++i) { aq[t][i] = al[i]; aq[t][8 + i] = ah[i]; }
    }

    float gq[8], m_r[8], l_r[8];
    #pragma unroll
    for (int r = 0; r < 8; ++r) {
        const int qq = q0 + r + hi * 8;
        gq[r] = gates[(size_t)(b * L + qq) * NB + blk];
        m_r[r] = NEGINF;
        l_r[r] = 0.0f;
    }
    v8f acc[4] = {};

    const int nch = qt / 2 + 1;          // 32-key chunks up to the diagonal
    for (int c = 0; c < nch; ++c) {
        const int k0 = c * 32;
        const int key0 = k0 + lo, key1 = k0 + 16 + lo;
        v8f S0 = {}, S1 = {};
        {
            const _Float16* kp0 = k + ((size_t)(b * L + key0)) * H + h * HD + hi * 16;
            const _Float16* kp1 = k + ((size_t)(b * L + key1)) * H + h * HD + hi * 16;
            #pragma unroll
            for (int t = 0; t < 2; ++t) {
                v16h b0 = *(const v16h*)(kp0 + t * 32);
                v16h b1 = *(const v16h*)(kp1 + t * 32);
                S0 = __builtin_amdgcn_wmma_f32_16x16x32_f16(
                    false, aq[t], false, b0, (short)0, S0, false, false);
                S1 = __builtin_amdgcn_wmma_f32_16x16x32_f16(
                    false, aq[t], false, b1, (short)0, S1, false, false);
            }
        }
        const float gk0 = gates[(size_t)(b * L + key0) * NB + blk];
        const float gk1 = gates[(size_t)(b * L + key1) * NB + blk];

        #pragma unroll
        for (int r = 0; r < 8; ++r) {
            const int qq = q0 + r + hi * 8;
            float v0 = (key0 <= qq) ? S0[r] * scale * gq[r] * gk0 : NEGINF;
            float v1 = (key1 <= qq) ? S1[r] * scale * gq[r] * gk1 : NEGINF;
            float rm = fmaxf(v0, v1);
            #pragma unroll
            for (int msk = 1; msk < 16; msk <<= 1)
                rm = fmaxf(rm, __shfl_xor(rm, msk, 32));
            const float mn = fmaxf(m_r[r], rm);
            const float corr = __expf(m_r[r] - mn);
            const float p0 = __expf(v0 - mn);
            const float p1 = __expf(v1 - mn);
            float rs = p0 + p1;
            #pragma unroll
            for (int msk = 1; msk < 16; msk <<= 1)
                rs += __shfl_xor(rs, msk, 32);
            l_r[r] = l_r[r] * corr + rs;
            m_r[r] = mn;
            #pragma unroll
            for (int j = 0; j < 4; ++j) acc[j][r] *= corr;
            Pst[(r + hi * 8) * 32 + lo]      = (_Float16)p0;
            Pst[(r + hi * 8) * 32 + 16 + lo] = (_Float16)p1;
        }
        __syncthreads();
        v16h ap;
        {
            v8h pl = *(const v8h*)(&Pst[lo * 32 + hi * 16]);
            v8h ph = *(const v8h*)(&Pst[lo * 32 + hi * 16 + 8]);
            #pragma unroll
            for (int i = 0; i < 8; ++i) { ap[i] = pl[i]; ap[8 + i] = ph[i]; }
        }
        __syncthreads();
        #pragma unroll
        for (int j = 0; j < 4; ++j) {
            const _Float16* vp =
                vT + ((size_t)((b * NH + h) * HD + j * 16 + lo)) * L + k0 + hi * 16;
            v16h bv = *(const v16h*)vp;
            acc[j] = __builtin_amdgcn_wmma_f32_16x16x32_f16(
                false, ap, false, bv, (short)0, acc[j], false, false);
        }
    }

    #pragma unroll
    for (int j = 0; j < 4; ++j) {
        #pragma unroll
        for (int r = 0; r < 8; ++r) {
            const int row = q0 + r + hi * 8;
            out[((size_t)(b * L + row)) * H + h * HD + j * 16 + lo] =
                acc[j][r] / l_r[r];
        }
    }
}

// ---------------------------------------------------------------------------
extern "C" void kernel_launch(void* const* d_in, const int* in_sizes, int n_in,
                              void* d_out, int out_size, void* d_ws, size_t ws_size,
                              hipStream_t stream)
{
    (void)in_sizes; (void)n_in; (void)out_size; (void)ws_size;
    const int B = 64, L = 512, H = 256, NB = 2, NH = 4, HD = 64;
    const int R = B * L;                       // 32768 rows
    const size_t E = (size_t)R * H;            // 8,388,608 elements
    const size_t WHH = (size_t)NB * H * H;     // per-weight elements

    const int*   seqs_data = (const int*)  d_in[0];
    const float* seqs      = (const float*)d_in[1];
    const int*   position  = (const int*)  d_in[2];
    const float* time_emb  = (const float*)d_in[3];
    const float* pos_table = (const float*)d_in[4];
    const float* gate_W    = (const float*)d_in[5];
    const float* gate_b    = (const float*)d_in[6];
    const float* ln_attn_g = (const float*)d_in[7];
    const float* ln_attn_b = (const float*)d_in[8];
    const float* qW        = (const float*)d_in[9];
    const float* qb        = (const float*)d_in[10];
    const float* kW        = (const float*)d_in[11];
    const float* kb        = (const float*)d_in[12];
    const float* vW        = (const float*)d_in[13];
    const float* vb        = (const float*)d_in[14];
    const float* ln_ffn_g  = (const float*)d_in[15];
    const float* ln_ffn_b  = (const float*)d_in[16];
    const float* c1W       = (const float*)d_in[17];
    const float* c1b       = (const float*)d_in[18];
    const float* c2W       = (const float*)d_in[19];
    const float* c2b       = (const float*)d_in[20];
    const float* last_g    = (const float*)d_in[21];
    const float* last_b    = (const float*)d_in[22];

    char* ws = (char*)d_ws;
    size_t off = 0;
    auto alloc = [&](size_t bytes) -> char* {
        char* p = ws + off;
        off += (bytes + 255) & ~(size_t)255;
        return p;
    };
    float*    s32   = (float*)   alloc(E * 4);   // working seqs
    float*    t32   = (float*)   alloc(E * 4);   // time
    float*    Q32   = (float*)   alloc(E * 4);   // LN(seqs)
    float*    x32   = (float*)   alloc(E * 4);   // FFN input
    float*    attno = (float*)   alloc(E * 4);   // attention output
    _Float16* t16   = (_Float16*)alloc(E * 2);
    _Float16* Q16   = (_Float16*)alloc(E * 2);
    _Float16* tmp16 = (_Float16*)alloc(E * 2);   // shared: (seqs+time) f16, then relu-h f16
    _Float16* q16   = (_Float16*)alloc(E * 2);
    _Float16* k16   = (_Float16*)alloc(E * 2);
    _Float16* v16   = (_Float16*)alloc(E * 2);
    _Float16* vT16  = (_Float16*)alloc(E * 2);
    _Float16* x16   = (_Float16*)alloc(E * 2);
    float*    gates = (float*)   alloc((size_t)R * NB * 4);
    float*    keep  = (float*)   alloc((size_t)R * 4);
    _Float16* qW16  = (_Float16*)alloc(WHH * 2);
    _Float16* kW16  = (_Float16*)alloc(WHH * 2);
    _Float16* vW16  = (_Float16*)alloc(WHH * 2);
    _Float16* c1W16 = (_Float16*)alloc(WHH * 2);
    _Float16* c2W16 = (_Float16*)alloc(WHH * 2);

    const dim3 gBlk(32, 4, 1);                 // 4 waves per GEMM block
    const int  gGrid = R / 16;                 // 2048 M-tiles
    const int  eGrid = (int)((E + 255) / 256);
    const int  wGrid = (int)((WHH + 255) / 256);

    prep_kernel<<<R, 256, 0, stream>>>(seqs_data, seqs, position, time_emb, pos_table,
                                       gate_W, gate_b, s32, t32, t16, gates, keep, H, NB);

    cvt_f32_f16<<<wGrid, 256, 0, stream>>>(qW,  qW16,  WHH);
    cvt_f32_f16<<<wGrid, 256, 0, stream>>>(kW,  kW16,  WHH);
    cvt_f32_f16<<<wGrid, 256, 0, stream>>>(vW,  vW16,  WHH);
    cvt_f32_f16<<<wGrid, 256, 0, stream>>>(c1W, c1W16, WHH);
    cvt_f32_f16<<<wGrid, 256, 0, stream>>>(c2W, c2W16, WHH);

    for (int i = 0; i < NB; ++i) {
        const size_t whh = (size_t)i * H * H;
        const size_t wh  = (size_t)i * H;

        // Q = LN(seqs)
        ln_kernel<<<R, 256, 0, stream>>>(s32, nullptr, ln_attn_g + wh, ln_attn_b + wh,
                                         Q32, Q16, H);
        // q = Q @ qW^T + qb
        gemm16_kernel<<<gGrid, gBlk, 0, stream>>>(Q16, qW16 + whh, qb + wh,
                                                  nullptr, nullptr, nullptr, q16,
                                                  R, H, H, 0);
        // k = time @ kW^T + kb
        gemm16_kernel<<<gGrid, gBlk, 0, stream>>>(t16, kW16 + whh, kb + wh,
                                                  nullptr, nullptr, nullptr, k16,
                                                  R, H, H, 0);
        // v = (seqs + time) @ vW^T + vb
        add_f32_to_f16<<<eGrid, 256, 0, stream>>>(s32, t32, tmp16, E);
        gemm16_kernel<<<gGrid, gBlk, 0, stream>>>(tmp16, vW16 + whh, vb + wh,
                                                  nullptr, nullptr, nullptr, v16,
                                                  R, H, H, 0);
        transpose_v_kernel<<<eGrid, 256, 0, stream>>>(v16, vT16, B, L, NH, HD);

        // gated causal flash attention
        attn_kernel<<<dim3(L / 16, NH, B), 32, 0, stream>>>(
            q16, k16, vT16, gates, attno, B, L, H, NH, HD, NB, i, 0.125f);

        // x = LN(Q + out)
        ln_kernel<<<R, 256, 0, stream>>>(attno, Q32, ln_ffn_g + wh, ln_ffn_b + wh,
                                         x32, x16, H);
        // h = relu(x @ c1W^T + c1b)
        gemm16_kernel<<<gGrid, gBlk, 0, stream>>>(x16, c1W16 + whh, c1b + wh,
                                                  nullptr, nullptr, nullptr, tmp16,
                                                  R, H, H, 1);
        // seqs = (h @ c2W^T + c2b + x) * keep
        gemm16_kernel<<<gGrid, gBlk, 0, stream>>>(tmp16, c2W16 + whh, c2b + wh,
                                                  x32, keep, s32, nullptr,
                                                  R, H, H, 0);
    }

    // final LN -> d_out (f32)
    ln_kernel<<<R, 256, 0, stream>>>(s32, nullptr, last_g, last_b,
                                     (float*)d_out, nullptr, H);
}
